// NewHyperLinear_59176059404509
// MI455X (gfx1250) — matrix operand order repaired and checked by reading, not compiled, not run on hardware
//
#include <hip/hip_runtime.h>
#include <hip/hip_bf16.h>

typedef _Float16 v16h __attribute__((ext_vector_type(16)));
typedef _Float16 v8h  __attribute__((ext_vector_type(8)));
typedef _Float16 v4h  __attribute__((ext_vector_type(4)));
typedef float    v8f  __attribute__((ext_vector_type(8)));

#define BATCH 512
#define INF   512
#define OUTF  512
#define NS    128
#define D0    (INF * OUTF)      /* 262144 */
#define KSPAN 128               /* i-range per k-slice */
#define NSP   136               /* 128 + 8 f16 pad (272B row stride) */
#define NSX   36                /* x-slice row stride: 72B, 8B-aligned, bank-clean */

union V16 { v16h v; v8h h[2]; };
union V8F { v8f v; float4 f[2]; };

__device__ __forceinline__ v16h hscale(v16h a, _Float16 s) {
  v16h r;
#pragma unroll
  for (int e = 0; e < 16; ++e) r[e] = a[e] * s;
  return r;
}

__device__ __forceinline__ v4h cvt4(float4 f) {
  v4h h; h[0] = (_Float16)f.x; h[1] = (_Float16)f.y; h[2] = (_Float16)f.z; h[3] = (_Float16)f.w;
  return h;
}

// Fused dual hyper-GEMM (operand-swapped: weights = A, x*noise = B):
// partial[z][b][o] = sum over i in slice z of
//   x[b,i]*noise[b,n]*hW[(i*512+o),n] + px[b,i]*noise[b,n]*pW[(i*512+o),n]
// + x[b,i]*hb[i*512+o] + px[b,i]*pb[i*512+o]
// Grid: blockIdx.x = batch tile (fast-varying -> co-scheduled WGs share the
// same weight stream in L2), blockIdx.y = output tile, blockIdx.z = k-slice.
__global__ __launch_bounds__(256) void hyper_gemm(
    const float* __restrict__ x,  const float* __restrict__ px,
    const float* __restrict__ noise,
    const float* __restrict__ hW, const float* __restrict__ pW,
    const float* __restrict__ hb, const float* __restrict__ pb,
    float* __restrict__ part)
{
  __shared__ _Float16 sB[2][64][NSP];   // [mat][o][n] f16, ~34.8 KB
  __shared__ _Float16 sX[2][64][NSX];   // [mat][batch_row][i%32] f16, ~9.2 KB

  const int t     = threadIdx.x;
  const int wave  = t >> 5;
  const int lane  = t & 31;
  const int lm    = lane & 15;
  const int hi    = lane >> 4;          // lane >= 16
  const int koff  = hi ? 8 : 0;         // A-fragment K sub-offset
  const int khalf = hi ? 16 : 0;        // unused; B fragments also use koff

  const int m0 = blockIdx.x * 64;       // batch tile  (N of the WMMA) - fast axis
  const int o0 = blockIdx.y * 64;       // output tile (M of the WMMA)
  const int z  = blockIdx.z;            // k-slice
  const int i0 = z * KSPAN;

  const int wave_o = wave & 1;          // 2 strips of 32 output rows
  const int wave_b = wave >> 1;         // 4 strips of 16 batch cols
  const int orow0  = wave_o * 32 + lm;  // A row for osub 0 (osub 1 = +16)
  const int brow   = wave_b * 16 + lm;  // local batch row (sX row, B column)
  const int gb     = m0 + brow;         // global batch row

  // ---- resident noise B-fragments (i-invariant): lane=batch col, n at koff+0..7, koff+16..23 ----
  v16h nb[4];
  {
    const float* nr = noise + (size_t)gb * NS;
#pragma unroll
    for (int s = 0; s < 4; ++s) {
      const float4* p = (const float4*)(nr + 32 * s + koff);
#pragma unroll
      for (int q = 0; q < 4; ++q) {
        float4 f = p[q < 2 ? q : q + 2];
        nb[s][4 * q + 0] = (_Float16)f.x; nb[s][4 * q + 1] = (_Float16)f.y;
        nb[s][4 * q + 2] = (_Float16)f.z; nb[s][4 * q + 3] = (_Float16)f.w;
      }
    }
  }

  v8f acc[2] = {};

  // ---- software-pipelined staging of contiguous 64x128 f32 weight slabs ----
  float4 pre[16];
  auto do_prefetch = [&](int i) {
    const float4* s0 = (const float4*)(hW + ((size_t)i * 512 + o0) * 128);
    const float4* s1 = (const float4*)(pW + ((size_t)i * 512 + o0) * 128);
#pragma unroll
    for (int j = 0; j < 8; ++j) { pre[j] = s0[j * 256 + t]; pre[8 + j] = s1[j * 256 + t]; }
  };
  auto do_store = [&]() {
    const int r = t >> 5;
    const int c = (t & 31) * 4;
#pragma unroll
    for (int j = 0; j < 8; ++j) {
      *(v4h*)&sB[0][j * 8 + r][c] = cvt4(pre[j]);
      *(v4h*)&sB[1][j * 8 + r][c] = cvt4(pre[8 + j]);
    }
  };
  auto stage_x = [&](int ibase) {
    const int row = t >> 2;             // 0..63
    const int cb  = (t & 3) * 8;        // 0,8,16,24
    const float4* s0 = (const float4*)(x  + (size_t)(m0 + row) * 512 + ibase + cb);
    const float4* s1 = (const float4*)(px + (size_t)(m0 + row) * 512 + ibase + cb);
    float4 a0 = s0[0], a1 = s0[1], b0 = s1[0], b1 = s1[1];
    *(v4h*)&sX[0][row][cb]     = cvt4(a0);
    *(v4h*)&sX[0][row][cb + 4] = cvt4(a1);
    *(v4h*)&sX[1][row][cb]     = cvt4(b0);
    *(v4h*)&sX[1][row][cb + 4] = cvt4(b1);
  };

  do_prefetch(i0);
#pragma unroll 1
  for (int ii = 0; ii < KSPAN; ++ii) {
    const int i = i0 + ii;
    do_store();
    if ((ii & 31) == 0) stage_x(i);
    __syncthreads();
    if (ii + 1 < KSPAN) do_prefetch(i + 1);   // global loads fly under the WMMAs

    const int ixl = ii & 31;
    _Float16 xs0 = sX[0][brow][ixl];          // ds_u16, lane-broadcast friendly
    _Float16 xs1 = sX[1][brow][ixl];

    v16h bf[2][2];                            // [kstep parity][mat] scaled B frags
#pragma unroll
    for (int s = 0; s < 4; ++s) {
      const int p = s & 1;
      const int off = 32 * s;
      // A fragments: weight tiles straight from LDS (no VALU, no WMMA/VALU hazard)
      V16 ah0, ah1, ap0, ap1;
      ah0.h[0] = *(const v8h*)&sB[0][orow0][off + koff];
      ah0.h[1] = *(const v8h*)&sB[0][orow0][off + 16 + koff];
      ah1.h[0] = *(const v8h*)&sB[0][orow0 + 16][off + koff];
      ah1.h[1] = *(const v8h*)&sB[0][orow0 + 16][off + 16 + koff];
      ap0.h[0] = *(const v8h*)&sB[1][orow0][off + koff];
      ap0.h[1] = *(const v8h*)&sB[1][orow0][off + 16 + koff];
      ap1.h[0] = *(const v8h*)&sB[1][orow0 + 16][off + koff];
      ap1.h[1] = *(const v8h*)&sB[1][orow0 + 16][off + 16 + koff];
      // B fragments: shared across both o-subtiles
      bf[p][0] = hscale(nb[s], xs0);
      bf[p][1] = hscale(nb[s], xs1);
      acc[0] = __builtin_amdgcn_wmma_f32_16x16x32_f16(
          false, ah0.v, false, bf[p][0], (short)0, acc[0], false, false);
      acc[1] = __builtin_amdgcn_wmma_f32_16x16x32_f16(
          false, ah1.v, false, bf[p][0], (short)0, acc[1], false, false);
      acc[0] = __builtin_amdgcn_wmma_f32_16x16x32_f16(
          false, ap0.v, false, bf[p][1], (short)0, acc[0], false, false);
      acc[1] = __builtin_amdgcn_wmma_f32_16x16x32_f16(
          false, ap1.v, false, bf[p][1], (short)0, acc[1], false, false);
    }
    __syncthreads();
  }

  // ---- tail: x @ reshape(hb) + px @ reshape(pb) over this slice's i-range ----
#pragma unroll 1
  for (int c4 = 0; c4 < 4; ++c4) {
    const int ic = i0 + 32 * c4;
    {
      const int k  = t >> 3;            // 0..31 (i within chunk)
      const int o8 = (t & 7) * 8;       // 0..56
      const float* s0 = hb + (size_t)(ic + k) * 512 + o0 + o8;
      const float* s1 = pb + (size_t)(ic + k) * 512 + o0 + o8;
#pragma unroll
      for (int j = 0; j < 8; ++j) sB[0][o8 + j][k] = (_Float16)s0[j];
#pragma unroll
      for (int j = 0; j < 8; ++j) sB[1][o8 + j][k] = (_Float16)s1[j];
    }
    __syncthreads();
    {
      // B fragments from x / prior_x rows (f32 at koff+0..7, koff+16..23 per lane)
      v16h bx, bp;
      const float* xr = x  + (size_t)gb * 512 + ic + koff;
      const float* pr = px + (size_t)gb * 512 + ic + koff;
#pragma unroll
      for (int q = 0; q < 4; ++q) {
        float4 fx = ((const float4*)xr)[q < 2 ? q : q + 2], fp = ((const float4*)pr)[q < 2 ? q : q + 2];
        bx[4 * q + 0] = (_Float16)fx.x; bx[4 * q + 1] = (_Float16)fx.y;
        bx[4 * q + 2] = (_Float16)fx.z; bx[4 * q + 3] = (_Float16)fx.w;
        bp[4 * q + 0] = (_Float16)fp.x; bp[4 * q + 1] = (_Float16)fp.y;
        bp[4 * q + 2] = (_Float16)fp.z; bp[4 * q + 3] = (_Float16)fp.w;
      }
      V16 ah0, ah1, ap0, ap1;
      ah0.h[0] = *(const v8h*)&sB[0][orow0][koff];
      ah0.h[1] = *(const v8h*)&sB[0][orow0][16 + koff];
      ah1.h[0] = *(const v8h*)&sB[0][orow0 + 16][koff];
      ah1.h[1] = *(const v8h*)&sB[0][orow0 + 16][16 + koff];
      ap0.h[0] = *(const v8h*)&sB[1][orow0][koff];
      ap0.h[1] = *(const v8h*)&sB[1][orow0][16 + koff];
      ap1.h[0] = *(const v8h*)&sB[1][orow0 + 16][koff];
      ap1.h[1] = *(const v8h*)&sB[1][orow0 + 16][16 + koff];
      acc[0] = __builtin_amdgcn_wmma_f32_16x16x32_f16(
          false, ah0.v, false, bx, (short)0, acc[0], false, false);
      acc[1] = __builtin_amdgcn_wmma_f32_16x16x32_f16(
          false, ah1.v, false, bx, (short)0, acc[1], false, false);
      acc[0] = __builtin_amdgcn_wmma_f32_16x16x32_f16(
          false, ap0.v, false, bp, (short)0, acc[0], false, false);
      acc[1] = __builtin_amdgcn_wmma_f32_16x16x32_f16(
          false, ap1.v, false, bp, (short)0, acc[1], false, false);
    }
    __syncthreads();
  }

  // ---- write partial tile: D rows = contiguous o -> 2x float4 per lane per osub ----
  {
    float* dst = part + (size_t)z * (BATCH * OUTF) + (size_t)gb * OUTF;
#pragma unroll
    for (int m = 0; m < 2; ++m) {
      const int ob = o0 + wave_o * 32 + m * 16 + (hi ? 8 : 0);
      V8F u; u.v = acc[m];
      *(float4*)&dst[ob]     = u.f[0];
      *(float4*)&dst[ob + 4] = u.f[1];
    }
  }
}

// out[b,o] = sum_z partial[z][b,o] + noise[b,:]@(hW+pW)[D0+o,:] + hb[D0+o] + pb[D0+o]
__global__ __launch_bounds__(256) void hyper_reduce(
    const float* __restrict__ part, const float* __restrict__ noise,
    const float* __restrict__ hW, const float* __restrict__ pW,
    const float* __restrict__ hb, const float* __restrict__ pb,
    float* __restrict__ out)
{
  const int idx = blockIdx.x * 256 + threadIdx.x;   // 0..262143
  const int b = idx >> 9;
  const int o = idx & 511;
  const int P = BATCH * OUTF;
  float s = part[idx] + part[P + idx] + part[2 * P + idx] + part[3 * P + idx];
  const float4* hw4 = (const float4*)(hW + (size_t)(D0 + o) * NS);
  const float4* pw4 = (const float4*)(pW + (size_t)(D0 + o) * NS);
  const float4* nz4 = (const float4*)(noise + (size_t)b * NS);
#pragma unroll 4
  for (int j = 0; j < NS / 4; ++j) {
    float4 n = nz4[j], a = hw4[j], c = pw4[j];
    s += n.x * (a.x + c.x) + n.y * (a.y + c.y) + n.z * (a.z + c.z) + n.w * (a.w + c.w);
  }
  s += hb[D0 + o] + pb[D0 + o];
  out[idx] = s;
}

extern "C" void kernel_launch(void* const* d_in, const int* in_sizes, int n_in,
                              void* d_out, int out_size, void* d_ws, size_t ws_size,
                              hipStream_t stream) {
  (void)in_sizes; (void)n_in; (void)out_size; (void)ws_size;
  const float* x     = (const float*)d_in[0];
  const float* px    = (const float*)d_in[1];
  const float* noise = (const float*)d_in[2];
  const float* hW    = (const float*)d_in[3];
  const float* hb    = (const float*)d_in[4];
  const float* pW    = (const float*)d_in[5];
  const float* pb    = (const float*)d_in[6];
  float* out  = (float*)d_out;
  float* part = (float*)d_ws;   // 4 * 512 * 512 f32 = 4 MB

  dim3 grid(BATCH / 64, OUTF / 64, 4);   // batch tile on x: shared weight streams
  hyper_gemm<<<grid, 256, 0, stream>>>(x, px, noise, hW, pW, hb, pb, part);
  hyper_reduce<<<(BATCH * OUTF) / 256, 256, 0, stream>>>(part, noise, hW, pW, hb, pb, out);
}
